// Merge_MixtralSparseMoeBlock_86131274154100
// MI455X (gfx1250) — matrix operand order repaired and checked by reading, not compiled, not run on hardware
//
#include <hip/hip_runtime.h>

// ---------------------------------------------------------------------------
// MoE (Mixtral-style, low-rank expert deltas) for MI455X / gfx1250.
//
// f16 WMMA (v_wmma_f32_16x16x32_f16, f32 accum) everywhere; fp32 inputs
// converted once to f16 workspace.  128x128 block GEMM, double-buffered LDS
// fed by GLOBAL_LOAD_ASYNC_TO_LDS_B128 (ASYNCcnt), 8 WMMA per wave per
// k-step.  comb weights folded into hmid so the whole down path
// (shared Wm2 GEMM + per-expert low-rank delta) runs off one weighted
// hmid buffer.
// ---------------------------------------------------------------------------

typedef __attribute__((ext_vector_type(16))) _Float16 v16h;
typedef __attribute__((ext_vector_type(8)))  float    v8f;
typedef __attribute__((ext_vector_type(8)))  unsigned int v8u;

#define TT 1024      // tokens
#define HH 2048      // hidden
#define II 7168      // intermediate
#define EE 8         // experts
#define RR 159       // low rank
#define RP 160       // padded rank (5 WMMA k-steps of 32)

// ---- CDNA5 async copy + WMMA helpers --------------------------------------

__device__ __forceinline__ void async_load_b128(void* lds, const void* gaddr) {
  // Low 32 bits of a generic shared-space pointer == LDS byte address.
  unsigned l = (unsigned)(size_t)lds;
  asm volatile("global_load_async_to_lds_b128 %0, %1, off"
               :: "v"(l), "v"(gaddr) : "memory");
}

__device__ __forceinline__ void wait_async() {
  asm volatile("s_wait_asynccnt 0" ::: "memory");
}

__device__ __forceinline__ v8f wmma_f16(v16h a, v16h b, v8f c) {
  return __builtin_amdgcn_wmma_f32_16x16x32_f16(
      /*neg_a=*/false, a, /*neg_b=*/false, b,
      /*c_mod=*/(short)0, c, /*reuse_a=*/false, /*reuse_b=*/false);
}

// A fragment: 16x32 f16, row-major source (ld halfs, even).
__device__ __forceinline__ v16h frag_a(const _Float16* p0, int ld, int lane) {
  const int m = lane & 15, hi = (lane >> 4) & 1;
  const unsigned int* p = (const unsigned int*)(p0 + (size_t)m * ld);
  v8u t;
#pragma unroll
  for (int v = 0; v < 8; ++v) {
    int k = 2 * v + ((v >= 4) ? 8 : 0) + hi * 8;
    t[v] = p[k >> 1];
  }
  return __builtin_bit_cast(v16h, t);
}

// B fragment: 32x16 f16 (K x N); source stored N-major (row n holds K contig).
__device__ __forceinline__ v16h frag_b(const _Float16* p0, long ld, int lane) {
  const int n = lane & 15, hi = (lane >> 4) & 1;
  const unsigned int* p = (const unsigned int*)(p0 + (size_t)n * ld);
  v8u t;
#pragma unroll
  for (int v = 0; v < 8; ++v) t[v] = p[v + ((v >= 4) ? 4 : 0) + hi * 4];
  return __builtin_bit_cast(v16h, t);
}

// ---- small utility kernels ------------------------------------------------

__global__ __launch_bounds__(256) void init_kernel(int* cnt) {
  if (threadIdx.x < EE) cnt[threadIdx.x] = 0;
}

__global__ __launch_bounds__(256) void f32_to_f16_kernel(const float* __restrict__ s,
                                                         _Float16* __restrict__ d, size_t n) {
  size_t i = (size_t)blockIdx.x * 256 + threadIdx.x;
  if (i < n) d[i] = (_Float16)s[i];
}

// src [rows][RR] -> dst [rows][RP], pad with zeros
__global__ __launch_bounds__(256) void pad_inner_kernel(const float* __restrict__ s,
                                                        _Float16* __restrict__ d,
                                                        size_t rows) {
  size_t i = (size_t)blockIdx.x * 256 + threadIdx.x;
  size_t n = rows * RP;
  if (i >= n) return;
  int r = (int)(i % RP);
  size_t row = i / RP;
  d[i] = (r < RR) ? (_Float16)s[row * RR + r] : (_Float16)0.f;
}

// src [EE][RR][inner] -> dst [EE][RP][inner], pad rows with zeros
__global__ __launch_bounds__(256) void pad_mid_kernel(const float* __restrict__ s,
                                                      _Float16* __restrict__ d,
                                                      size_t inner) {
  size_t i = (size_t)blockIdx.x * 256 + threadIdx.x;
  size_t n = (size_t)EE * RP * inner;
  if (i >= n) return;
  size_t e = i / (RP * inner);
  size_t rem = i % (RP * inner);
  int r = (int)(rem / inner);
  size_t c = rem % inner;
  d[i] = (r < RR) ? (_Float16)s[(e * RR + r) * inner + c] : (_Float16)0.f;
}

// ---- router: logits, softmax, top-2, comb weights, expert lists, x->f16 ---

__global__ __launch_bounds__(256) void router_kernel(const float* __restrict__ x,
                                                     const float* __restrict__ gw,
                                                     float* __restrict__ logits,
                                                     float* __restrict__ comb,
                                                     int* __restrict__ cnt,
                                                     int* __restrict__ list,
                                                     _Float16* __restrict__ hs) {
  const int t = blockIdx.x, tid = threadIdx.x;
  float a[EE];
#pragma unroll
  for (int e = 0; e < EE; ++e) a[e] = 0.f;
  for (int h = tid; h < HH; h += 256) {
    float xv = x[(size_t)t * HH + h];
    hs[(size_t)t * HH + h] = (_Float16)xv;
#pragma unroll
    for (int e = 0; e < EE; ++e) a[e] += xv * gw[(size_t)e * HH + h];
  }
  __shared__ float red[256];
  __shared__ float lg[EE];
  for (int e = 0; e < EE; ++e) {
    red[tid] = a[e];
    __syncthreads();
    for (int s = 128; s > 0; s >>= 1) {
      if (tid < s) red[tid] += red[tid + s];
      __syncthreads();
    }
    if (tid == 0) lg[e] = red[0];
    __syncthreads();
  }
  if (tid == 0) {
    float m = lg[0];
#pragma unroll
    for (int e = 1; e < EE; ++e) m = fmaxf(m, lg[e]);
    float p[EE], sum = 0.f;
#pragma unroll
    for (int e = 0; e < EE; ++e) { p[e] = __expf(lg[e] - m); sum += p[e]; }
#pragma unroll
    for (int e = 0; e < EE; ++e) {
      p[e] /= sum;
      logits[(size_t)t * EE + e] = lg[e];
      comb[(size_t)t * EE + e] = 0.f;
    }
    int i0 = 0;
#pragma unroll
    for (int e = 1; e < EE; ++e) if (p[e] > p[i0]) i0 = e;
    int i1 = (i0 == 0) ? 1 : 0;
#pragma unroll
    for (int e = 0; e < EE; ++e) if (e != i0 && p[e] > p[i1]) i1 = e;
    float w0 = p[i0], w1 = p[i1], s2 = w0 + w1;
    w0 /= s2; w1 /= s2;
    comb[(size_t)t * EE + i0] = w0;
    comb[(size_t)t * EE + i1] = w1;
    int p0 = atomicAdd(&cnt[i0], 1); list[i0 * TT + p0] = t * 2;
    int p1 = atomicAdd(&cnt[i1], 1); list[i1 * TT + p1] = t * 2 + 1;
  }
}

// ---- 128x128 block GEMM: C[M,N] = A[M,K] * B^T (B stored N-major) ---------
// Double-buffered LDS fed by global_load_async_to_lds_b128.
// AM 0: A row-major [M][lda].  AM 1: A row t = f16 sum of rows 2t,2t+1.
// CM 0: f32 out.  CM 1: f16 out.

template <int CM, int AM>
__global__ __launch_bounds__(256) void gemm_kernel(const _Float16* __restrict__ A,
                                                   const _Float16* __restrict__ B,
                                                   float* __restrict__ Cf,
                                                   _Float16* __restrict__ Ch,
                                                   long lda, long ldb, long ldc,
                                                   int K) {
  // 40-half row stride: rows are 16B aligned for async b128, and the
  // 20-bank stride walks all 16 fragment rows onto distinct banks.
  __shared__ _Float16 As[2][128][40];
  __shared__ _Float16 Bs[2][128][40];
  const int tid = threadIdx.x;
  const int lane = tid & 31, wid = tid >> 5;
  const int wr = wid & 3, wc = wid >> 2;
  const long m0 = (long)blockIdx.y * 128;
  const long n0 = (long)blockIdx.x * 128;

  auto stage = [&](int buf, int k0) {
#pragma unroll
    for (int it = 0; it < 2; ++it) {
      int idx = tid + it * 256;      // 512 chunks of 16B cover 128x32 halfs
      int r = idx >> 2, c = idx & 3;
      async_load_b128(&Bs[buf][r][c * 8], B + (n0 + r) * ldb + k0 + c * 8);
    }
    if (AM == 0) {
#pragma unroll
      for (int it = 0; it < 2; ++it) {
        int idx = tid + it * 256;
        int r = idx >> 2, c = idx & 3;
        async_load_b128(&As[buf][r][c * 8], A + (m0 + r) * lda + k0 + c * 8);
      }
    } else {
      for (int idx = tid; idx < 128 * 32; idx += 256) {
        int r = idx >> 5, c = idx & 31;
        const _Float16* p = A + (2 * (m0 + r)) * lda + k0 + c;
        As[buf][r][c] = (_Float16)((float)p[0] + (float)p[lda]);
      }
    }
  };

  v8f acc[2][4] = {};
  stage(0, 0);
  for (int k0 = 0; k0 < K; k0 += 32) {
    const int buf = (k0 >> 5) & 1;
    wait_async();
    __syncthreads();                      // buf ready; buf^1's last readers done
    if (k0 + 32 < K) stage(buf ^ 1, k0 + 32);
    v16h a0 = frag_a(&As[buf][wr * 32][0], 40, lane);
    v16h a1 = frag_a(&As[buf][wr * 32 + 16][0], 40, lane);
#pragma unroll
    for (int q = 0; q < 4; ++q) {
      v16h b = frag_b(&Bs[buf][wc * 64 + q * 16][0], 40, lane);
      acc[0][q] = wmma_f16(a0, b, acc[0][q]);
      acc[1][q] = wmma_f16(a1, b, acc[1][q]);
    }
  }
  const int n = lane & 15, hi = lane >> 4;
#pragma unroll
  for (int i = 0; i < 2; ++i) {
#pragma unroll
    for (int q = 0; q < 4; ++q) {
      const long col = n0 + wc * 64 + q * 16 + n;
#pragma unroll
      for (int j = 0; j < 8; ++j) {
        const long row = m0 + wr * 32 + i * 16 + hi * 8 + j;
        if (CM == 0) Cf[row * ldc + col] = acc[i][q][j];
        else         Ch[row * ldc + col] = (_Float16)acc[i][q][j];
      }
    }
  }
}

// ---- fused per-expert middle kernel ---------------------------------------
// For one (expert e, 16-token tile from list_e):
//   gate = base_gate + t1_e * du1_e^T ; up = base_up + t3_e * du3_e^T (WMMA)
//   whmid[t,slot] = comb * silu(gate) * up  (f16 scatter; weight folded here
//   so both down-path GEMMs read the same buffer)

__global__ __launch_bounds__(256) void middle_kernel(
    const _Float16* __restrict__ t1b, const _Float16* __restrict__ t3b,
    const _Float16* __restrict__ du1p, const _Float16* __restrict__ du3p,
    const float* __restrict__ base_gate, const float* __restrict__ base_up,
    const float* __restrict__ comb, const int* __restrict__ cnt,
    const int* __restrict__ list,
    _Float16* __restrict__ whmid) {
  const int e = blockIdx.y;
  const int rt = blockIdx.x;
  const int count = cnt[e];
  if (rt * 16 >= count) return;
  const int tid = threadIdx.x, lane = tid & 31, wid = tid >> 5;
  const int n = lane & 15, hi = lane >> 4;

  __shared__ int tok[16];
  __shared__ int slj[16];
  __shared__ float wgt[16];
  __shared__ int act[16];
  __shared__ _Float16 t1s[16][168];   // 336B rows: 16B aligned, 20-bank walk
  __shared__ _Float16 t3s[16][168];

  if (tid < 16) {
    int row = rt * 16 + tid;
    int a = (row < count) ? 1 : 0;
    int val = a ? list[e * TT + row] : 0;
    tok[tid] = val >> 1;
    slj[tid] = val & 1;
    wgt[tid] = a ? comb[(size_t)(val >> 1) * EE + e] : 0.f;
    act[tid] = a;
  }
  __syncthreads();
  // gathered rows of t1/t3 (160 halfs = 20 b128 chunks each), async to LDS
  for (int idx = tid; idx < 16 * 20; idx += 256) {
    int r = idx / 20, c = idx % 20;
    const size_t rowoff = (size_t)tok[r] * (EE * RP) + e * RP + c * 8;
    async_load_b128(&t1s[r][c * 8], t1b + rowoff);
    async_load_b128(&t3s[r][c * 8], t3b + rowoff);
  }
  wait_async();
  __syncthreads();

  const _Float16* du1e = du1p + (size_t)e * II * RP;
  const _Float16* du3e = du3p + (size_t)e * II * RP;

  for (int ic = wid * 32; ic < II; ic += 256) {
    v8f g[2] = {}, u[2] = {};
#pragma unroll
    for (int r0 = 0; r0 < RP; r0 += 32) {
      v16h a1 = frag_a(&t1s[0][r0], 168, lane);
      v16h a3 = frag_a(&t3s[0][r0], 168, lane);
#pragma unroll
      for (int sub = 0; sub < 2; ++sub) {
        v16h bg = frag_b(du1e + (size_t)(ic + sub * 16) * RP + r0, RP, lane);
        v16h bu = frag_b(du3e + (size_t)(ic + sub * 16) * RP + r0, RP, lane);
        g[sub] = wmma_f16(a1, bg, g[sub]);
        u[sub] = wmma_f16(a3, bu, u[sub]);
      }
    }
#pragma unroll
    for (int sub = 0; sub < 2; ++sub) {
      int col = ic + sub * 16 + n;
#pragma unroll
      for (int j = 0; j < 8; ++j) {
        int m = j + hi * 8;
        if (act[m]) {
          float gate = g[sub][j] + base_gate[(size_t)tok[m] * II + col];
          float up   = u[sub][j] + base_up[(size_t)tok[m] * II + col];
          float sv = gate / (1.f + __expf(-gate));
          whmid[((size_t)tok[m] * 2 + slj[m]) * II + col] =
              (_Float16)(wgt[m] * sv * up);
        }
      }
    }
  }
}

// ---- per-expert t2 GEMM: t2[e,row,:] = whmid_row x dv2_e^T ----------------
// A rows gathered via per-lane row pointers (fixed for whole K loop);
// 8 waves split K, deterministic wave-ordered reduction.

__global__ __launch_bounds__(256) void t2_kernel(const _Float16* __restrict__ whmid,
                                                 const _Float16* __restrict__ dv2p,
                                                 const int* __restrict__ cnt,
                                                 const int* __restrict__ list,
                                                 _Float16* __restrict__ t2b) {
  const int e = blockIdx.y;
  const int rt = blockIdx.x;
  const int count = cnt[e];
  if (rt * 16 >= count) return;
  const int tid = threadIdx.x, lane = tid & 31, wid = tid >> 5;
  const int n = lane & 15, hi = lane >> 4;
  __shared__ int rowid[16];
  __shared__ float t2a[16][160];
  if (tid < 16) {
    int row = rt * 16 + tid;
    rowid[tid] = (row < count) ? list[e * TT + row] : 0;   // tok*2+slot
  }
  for (int idx = tid; idx < 16 * 160; idx += 256) t2a[idx / 160][idx % 160] = 0.f;
  __syncthreads();

  const unsigned int* arow =
      (const unsigned int*)(whmid + (size_t)rowid[lane & 15] * II);
  const _Float16* dv2e = dv2p + (size_t)e * RP * II;

  v8f acc[10] = {};
  for (int k0 = wid * 32; k0 < II; k0 += 256) {
    v8u ta;
#pragma unroll
    for (int v = 0; v < 8; ++v) {
      int k = 2 * v + ((v >= 4) ? 8 : 0) + hi * 8 + k0;
      ta[v] = arow[k >> 1];
    }
    v16h a = __builtin_bit_cast(v16h, ta);
#pragma unroll
    for (int q = 0; q < 10; ++q) {
      v16h b = frag_b(dv2e + (size_t)(q * 16) * II + k0, (long)II, lane);
      acc[q] = wmma_f16(a, b, acc[q]);
    }
  }
  for (int w2 = 0; w2 < 8; ++w2) {
    if (wid == w2) {
#pragma unroll
      for (int q = 0; q < 10; ++q)
#pragma unroll
        for (int j = 0; j < 8; ++j)
          t2a[j + hi * 8][q * 16 + n] += acc[q][j];
    }
    __syncthreads();
  }
  for (int idx = tid; idx < 16 * 160; idx += 256) {
    int m = idx / 160, r = idx % 160;
    t2b[((size_t)e * TT + rt * 16 + m) * RP + r] = (_Float16)t2a[m][r];
  }
}

// ---- per-expert delta down-proj: delta[t,slot,:] = t2 * du2_e^T -----------
// (comb weight already folded into whmid -> t2)

__global__ __launch_bounds__(256) void ddown_kernel(const _Float16* __restrict__ t2b,
                                                    const _Float16* __restrict__ du2p,
                                                    const int* __restrict__ cnt,
                                                    const int* __restrict__ list,
                                                    float* __restrict__ delta) {
  const int e = blockIdx.z;
  const int rt = blockIdx.y;
  const int count = cnt[e];
  if (rt * 16 >= count) return;
  const int tid = threadIdx.x, lane = tid & 31, wid = tid >> 5;
  __shared__ _Float16 As[16][168];
  for (int idx = tid; idx < 16 * 20; idx += 256) {
    int r = idx / 20, c = idx % 20;
    async_load_b128(&As[r][c * 8],
                    t2b + ((size_t)e * TT + rt * 16 + r) * RP + c * 8);
  }
  wait_async();
  __syncthreads();
  const _Float16* du2e = du2p + (size_t)e * HH * RP;
  const int n0 = blockIdx.x * 128 + wid * 16;
  v8f acc = {};
#pragma unroll
  for (int r0 = 0; r0 < RP; r0 += 32) {
    v16h a = frag_a(&As[0][r0], 168, lane);
    v16h b = frag_b(du2e + (size_t)n0 * RP + r0, RP, lane);
    acc = wmma_f16(a, b, acc);
  }
  const int n = lane & 15, hi = lane >> 4;
#pragma unroll
  for (int j = 0; j < 8; ++j) {
    int m = j + hi * 8;
    int row = rt * 16 + m;
    if (row < count) {
      int val = list[e * TT + row];
      int t = val >> 1, js = val & 1;
      delta[((size_t)t * 2 + js) * HH + n0 + n] = acc[j];
    }
  }
}

// ---- final combine --------------------------------------------------------

__global__ __launch_bounds__(256) void combine_kernel(const float* __restrict__ base,
                                                      const float* __restrict__ delta,
                                                      float* __restrict__ out) {
  size_t i = (size_t)blockIdx.x * 256 + threadIdx.x;
  if (i >= (size_t)TT * HH) return;
  size_t t = i >> 11;          // /HH
  size_t h = i & (HH - 1);
  out[i] = base[i] + delta[(2 * t) * HH + h] + delta[(2 * t + 1) * HH + h];
}

// ---------------------------------------------------------------------------

extern "C" void kernel_launch(void* const* d_in, const int* in_sizes, int n_in,
                              void* d_out, int out_size, void* d_ws, size_t ws_size,
                              hipStream_t stream) {
  const float* x     = (const float*)d_in[0];
  const float* gw    = (const float*)d_in[1];
  const float* Wm1   = (const float*)d_in[2];
  const float* Wm2   = (const float*)d_in[3];
  const float* Wm3   = (const float*)d_in[4];
  const float* du1   = (const float*)d_in[5];
  const float* dv1   = (const float*)d_in[6];
  const float* du2   = (const float*)d_in[7];
  const float* dv2   = (const float*)d_in[8];
  const float* du3   = (const float*)d_in[9];
  const float* dv3   = (const float*)d_in[10];
  float* out = (float*)d_out;
  float* logits_out = out + (size_t)TT * HH;

  char* w = (char*)d_ws;
  auto alloc = [&](size_t bytes) -> char* {
    char* p = w;
    w += (bytes + 255) & ~(size_t)255;
    return p;
  };
  _Float16* hs    = (_Float16*)alloc((size_t)TT * HH * 2);
  _Float16* Wm1b  = (_Float16*)alloc((size_t)II * HH * 2);
  _Float16* Wm3b  = (_Float16*)alloc((size_t)II * HH * 2);
  _Float16* Wm2b  = (_Float16*)alloc((size_t)HH * II * 2);
  _Float16* dv1p  = (_Float16*)alloc((size_t)EE * RP * HH * 2);
  _Float16* dv3p  = (_Float16*)alloc((size_t)EE * RP * HH * 2);
  _Float16* dv2p  = (_Float16*)alloc((size_t)EE * RP * II * 2);
  _Float16* du1p  = (_Float16*)alloc((size_t)EE * II * RP * 2);
  _Float16* du3p  = (_Float16*)alloc((size_t)EE * II * RP * 2);
  _Float16* du2p  = (_Float16*)alloc((size_t)EE * HH * RP * 2);
  float*    bgate = (float*)alloc((size_t)TT * II * 4);
  float*    bup   = (float*)alloc((size_t)TT * II * 4);
  _Float16* t1b   = (_Float16*)alloc((size_t)TT * EE * RP * 2);
  _Float16* t3b   = (_Float16*)alloc((size_t)TT * EE * RP * 2);
  _Float16* whmid = (_Float16*)alloc((size_t)2 * TT * II * 2);
  _Float16* t2b   = (_Float16*)alloc((size_t)EE * TT * RP * 2);
  float*    obase = (float*)alloc((size_t)TT * HH * 4);
  float*    delta = (float*)alloc((size_t)2 * TT * HH * 4);
  float*    comb  = (float*)alloc((size_t)TT * EE * 4);
  int*      cnt   = (int*)alloc(EE * 4);
  int*      list  = (int*)alloc((size_t)EE * TT * 4);

  auto G = [](size_t n) { return (unsigned int)((n + 255) / 256); };

  init_kernel<<<1, 256, 0, stream>>>(cnt);
  router_kernel<<<TT, 256, 0, stream>>>(x, gw, logits_out, comb, cnt, list, hs);

  f32_to_f16_kernel<<<G((size_t)II * HH), 256, 0, stream>>>(Wm1, Wm1b, (size_t)II * HH);
  f32_to_f16_kernel<<<G((size_t)II * HH), 256, 0, stream>>>(Wm3, Wm3b, (size_t)II * HH);
  f32_to_f16_kernel<<<G((size_t)HH * II), 256, 0, stream>>>(Wm2, Wm2b, (size_t)HH * II);
  pad_mid_kernel<<<G((size_t)EE * RP * HH), 256, 0, stream>>>(dv1, dv1p, (size_t)HH);
  pad_mid_kernel<<<G((size_t)EE * RP * HH), 256, 0, stream>>>(dv3, dv3p, (size_t)HH);
  pad_mid_kernel<<<G((size_t)EE * RP * II), 256, 0, stream>>>(dv2, dv2p, (size_t)II);
  pad_inner_kernel<<<G((size_t)EE * II * RP), 256, 0, stream>>>(du1, du1p, (size_t)EE * II);
  pad_inner_kernel<<<G((size_t)EE * II * RP), 256, 0, stream>>>(du3, du3p, (size_t)EE * II);
  pad_inner_kernel<<<G((size_t)EE * HH * RP), 256, 0, stream>>>(du2, du2p, (size_t)EE * HH);

  // base gate / up:  [T,I] = hs[T,H] x Wm{1,3}[I,H]^T
  gemm_kernel<0, 0><<<dim3(II / 128, TT / 128), 256, 0, stream>>>(
      hs, Wm1b, bgate, nullptr, HH, HH, II, HH);
  gemm_kernel<0, 0><<<dim3(II / 128, TT / 128), 256, 0, stream>>>(
      hs, Wm3b, bup, nullptr, HH, HH, II, HH);

  // t1/t3: [T, E*RP] = hs x dv{1,3}p^T  (dense over experts; cheap)
  gemm_kernel<1, 0><<<dim3((EE * RP) / 128, TT / 128), 256, 0, stream>>>(
      hs, dv1p, nullptr, t1b, HH, HH, EE * RP, HH);
  gemm_kernel<1, 0><<<dim3((EE * RP) / 128, TT / 128), 256, 0, stream>>>(
      hs, dv3p, nullptr, t3b, HH, HH, EE * RP, HH);

  // per-expert middle: delta expansion + silu*up + weighted scatter
  middle_kernel<<<dim3(TT / 16, EE), 256, 0, stream>>>(
      t1b, t3b, du1p, du3p, bgate, bup, comb, cnt, list, whmid);

  // per-expert t2 = whmid x dv2^T (weight folded in)
  t2_kernel<<<dim3(TT / 16, EE), 256, 0, stream>>>(whmid, dv2p, cnt, list, t2b);

  // shared down-proj on comb-weighted hmid sum: [T,H] = whmid_sum x Wm2^T
  gemm_kernel<0, 1><<<dim3(HH / 128, TT / 128), 256, 0, stream>>>(
      whmid, Wm2b, obase, nullptr, II, II, HH, II);

  // per-expert low-rank down delta
  ddown_kernel<<<dim3(HH / 128, TT / 16, EE), 256, 0, stream>>>(
      t2b, du2p, cnt, list, delta);

  combine_kernel<<<G((size_t)TT * HH), 256, 0, stream>>>(obase, delta, out);
}
